// MarioDoomPolicyNet_12824772346597
// MI455X (gfx1250) — matrix operand order repaired and checked by reading, not compiled, not run on hardware
//
#include <hip/hip_runtime.h>

// ---------------- constants ----------------
#define TT 64
#define BB 64
#define HWH 42
#define CIN 4
#define NA 12
#define HID 256
#define FEAT 288
#define NB (TT*BB)          // 4096 images / rows

typedef __attribute__((ext_vector_type(16))) _Float16 v16h;
typedef __attribute__((ext_vector_type(8)))  float    v8f;

// ---------------- WMMA fragment helpers (wave32, gfx1250 layouts) ----------------
// A fragment: 16x32 f16 tile from row-major matrix, leading dim ld.
// lane l: row r=l&15, group g=l>>4; elems 0..7 = M[r][8g..8g+7], elems 8..15 = M[r][16+8g..]
__device__ __forceinline__ v16h load_a16(const _Float16* base, int ld) {
  const int l = threadIdx.x & 31;
  const int r = l & 15, g = l >> 4;
  const _Float16* p = base + r * ld + g * 8;
  v16h f;
#pragma unroll
  for (int i = 0; i < 8; ++i) { f[i] = p[i]; f[8 + i] = p[16 + i]; }
  return f;
}

// B fragment for B = W^T (so GEMM computes X @ W^T): W row-major (N x K), tile at W[n0][k0].
// lane l: column n = n0 + (l&15), g=l>>4; elems 0..7 = W[n][k0+8g..8g+7], elems 8..15 = W[n][k0+16+8g..].
__device__ __forceinline__ v16h load_b16(const _Float16* base, int ld) {
  const int l = threadIdx.x & 31;
  const int n = l & 15, g = l >> 4;
  const _Float16* p = base + n * ld + g * 8;
  v16h f;
#pragma unroll
  for (int i = 0; i < 8; ++i) { f[i] = p[i]; f[8 + i] = p[16 + i]; }
  return f;
}

__device__ __forceinline__ v8f wmma16(v16h a, v16h b, v8f c) {
  return __builtin_amdgcn_wmma_f32_16x16x32_f16(false, a, false, b, (short)0, c, false, false);
}

__device__ __forceinline__ float sigmf(float x) { return 1.0f / (1.0f + __expf(-x)); }
__device__ __forceinline__ float eluf(float x)  { return x > 0.0f ? x : expm1f(x); }

// ---------------- f32 -> f16 weight conversion ----------------
__global__ void cvt_f16_kernel(const float* __restrict__ src, _Float16* __restrict__ dst, int n) {
  int i = blockIdx.x * blockDim.x + threadIdx.x;
  if (i < n) dst[i] = (_Float16)src[i];
}

// pack policy(12x256)+baseline(1x256)+zeros into 16x256 f16 + 16-wide f32 bias
__global__ void head_pack_kernel(const float* __restrict__ pw, const float* __restrict__ pb,
                                 const float* __restrict__ bw, const float* __restrict__ bb,
                                 _Float16* __restrict__ hw, float* __restrict__ hb) {
  int i = blockIdx.x * blockDim.x + threadIdx.x;
  if (i < 16 * HID) {
    int r = i / HID, c = i - r * HID;
    float v = (r < NA) ? pw[r * HID + c] : ((r == NA) ? bw[c] : 0.0f);
    hw[i] = (_Float16)v;
  }
  if (i < 16) hb[i] = (i < NA) ? pb[i] : ((i == NA) ? bb[0] : 0.0f);
}

// ---------------- conv1: direct fp32 (K=36 too small for WMMA) ----------------
// x[n][c][y][x'] = frame[n][x'][y][c]; stride2 pad1; out 21x21x32, ELU, f16 out [n][oc][p]
__global__ void conv1_kernel(const float* __restrict__ frame, const float* __restrict__ w,
                             const float* __restrict__ b, _Float16* __restrict__ act1) {
  int idx = blockIdx.x * blockDim.x + threadIdx.x;
  const int NOUT = NB * 32 * 441;
  if (idx >= NOUT) return;
  int p  = idx % 441;
  int oc = (idx / 441) % 32;
  int n  = idx / (441 * 32);
  int oy = p / 21, ox = p % 21;
  float s = b[oc];
#pragma unroll
  for (int ky = 0; ky < 3; ++ky) {
    int iy = 2 * oy - 1 + ky;
    if (iy < 0 || iy >= HWH) continue;
#pragma unroll
    for (int kx = 0; kx < 3; ++kx) {
      int ix = 2 * ox - 1 + kx;
      if (ix < 0 || ix >= HWH) continue;
      const float* fr = frame + ((size_t)(n * HWH + ix) * HWH + iy) * CIN;
#pragma unroll
      for (int ic = 0; ic < CIN; ++ic)
        s += fr[ic] * w[((oc * CIN + ic) * 3 + ky) * 3 + kx];
    }
  }
  act1[((size_t)n * 32 + oc) * 441 + p] = (_Float16)eluf(s);
}

// ---------------- conv2..4: im2col in LDS + WMMA GEMM (M=pixels, N=32, K=288) ----------------
template <int INH, int OUTH, int MPAD>
__global__ void conv_wmma_kernel(const _Float16* __restrict__ actin,
                                 const _Float16* __restrict__ wgt,   // 32 x 288 f16
                                 const float* __restrict__ bias,     // 32 f32
                                 _Float16* __restrict__ actout) {    // [N][32][OUTP]
  constexpr int OUTP = OUTH * OUTH;
  constexpr int K = 288;
  extern __shared__ char dynsm[];
  _Float16* Alds = (_Float16*)dynsm;            // MPAD x 288
  const int n = blockIdx.x;
  const int tid = threadIdx.x;
  const _Float16* img = actin + (size_t)n * 32 * INH * INH;

  for (int idx = tid; idx < MPAD * K; idx += blockDim.x) {
    int row = idx / K, k = idx - row * K;
    _Float16 v = (_Float16)0.0f;
    if (row < OUTP) {
      int oy = row / OUTH, ox = row - oy * OUTH;
      int ic = k / 9, rr = k - ic * 9;
      int ky = rr / 3, kx = rr - ky * 3;
      int iy = 2 * oy - 1 + ky, ix = 2 * ox - 1 + kx;
      if (iy >= 0 && iy < INH && ix >= 0 && ix < INH)
        v = img[(ic * INH + iy) * INH + ix];
    }
    Alds[idx] = v;
  }
  __syncthreads();

  const int nwaves = blockDim.x >> 5;
  const int wid = tid >> 5;
  const int lane = tid & 31, r = lane & 15, g = lane >> 4;
  constexpr int NTILES = (MPAD / 16) * 2;
  for (int tile = wid; tile < NTILES; tile += nwaves) {
    int mt = tile >> 1, nt = tile & 1;
    v8f acc;
#pragma unroll
    for (int v = 0; v < 8; ++v) acc[v] = 0.0f;
    for (int ks = 0; ks < K / 32; ++ks) {
      v16h a = load_a16(Alds + mt * 16 * K + ks * 32, K);
      v16h b = load_b16(wgt + (nt * 16) * K + ks * 32, K);
      acc = wmma16(a, b, acc);
    }
    int oc = nt * 16 + r;
    float bv = bias[oc];
#pragma unroll
    for (int v = 0; v < 8; ++v) {
      int m = mt * 16 + v + 8 * g;
      if (m < OUTP)
        actout[((size_t)n * 32 + oc) * OUTP + m] = (_Float16)eluf(acc[v] + bv);
    }
  }
}

// ---------------- persistent single-WG LSTM (32 waves, state in LDS) ----------------
// Each wave owns 2 h-tiles (16 rows x 16 cols of h); for each it computes the 4 gate
// tiles (cols j, 256+j, 512+j, 768+j) with WMMA and fuses the LSTM nonlinearity.
template <int KX>
__device__ __forceinline__ void lstm_layer(
    const _Float16* __restrict__ Xmat, int ldx,       // 64 x KX input (global or LDS)
    const _Float16* hprev, float* cstate, _Float16* hout,
    const _Float16* __restrict__ wih, const _Float16* __restrict__ whh,
    const float* __restrict__ bih, const float* __restrict__ bhh,
    float* houtF, float* coutF, _Float16* seqout) {
  const int lane = threadIdx.x & 31;
  const int r = lane & 15, g = lane >> 4;
  float hn[2][8], cn[2][8];
#pragma unroll
  for (int s = 0; s < 2; ++s) {
    int ht = (threadIdx.x >> 5) + 32 * s;   // 0..63 over 4 mtiles x 16 coltiles
    int mt = ht & 3, ct = ht >> 2;
    int j = ct * 16 + r;
    v8f acc[4];
#pragma unroll
    for (int gg = 0; gg < 4; ++gg) {
      float bv = bih[gg * HID + j] + bhh[gg * HID + j];
#pragma unroll
      for (int v = 0; v < 8; ++v) acc[gg][v] = bv;
    }
    for (int ks = 0; ks < KX / 32; ++ks) {            // X @ W_ih^T
      v16h a = load_a16(Xmat + (mt * 16) * ldx + ks * 32, ldx);
#pragma unroll
      for (int gg = 0; gg < 4; ++gg) {
        v16h b = load_b16(wih + (gg * HID + ct * 16) * KX + ks * 32, KX);
        acc[gg] = wmma16(a, b, acc[gg]);
      }
    }
    for (int ks = 0; ks < HID / 32; ++ks) {           // H @ W_hh^T
      v16h a = load_a16(hprev + (mt * 16) * HID + ks * 32, HID);
#pragma unroll
      for (int gg = 0; gg < 4; ++gg) {
        v16h b = load_b16(whh + (gg * HID + ct * 16) * HID + ks * 32, HID);
        acc[gg] = wmma16(a, b, acc[gg]);
      }
    }
#pragma unroll
    for (int v = 0; v < 8; ++v) {
      int brow = mt * 16 + v + 8 * g;
      float iv = sigmf(acc[0][v]);
      float fv = sigmf(acc[1][v]);
      float gv = tanhf(acc[2][v]);
      float ov = sigmf(acc[3][v]);
      float cold = cstate[brow * HID + j];
      float cnew = fv * cold + iv * gv;
      cn[s][v] = cnew;
      hn[s][v] = ov * tanhf(cnew);
    }
  }
  __syncthreads();   // all reads of hprev done before anyone writes new h
#pragma unroll
  for (int s = 0; s < 2; ++s) {
    int ht = (threadIdx.x >> 5) + 32 * s;
    int mt = ht & 3, ct = ht >> 2;
    int j = ct * 16 + r;
#pragma unroll
    for (int v = 0; v < 8; ++v) {
      int brow = mt * 16 + v + 8 * g;
      hout[brow * HID + j] = (_Float16)hn[s][v];
      cstate[brow * HID + j] = cn[s][v];
      if (houtF) houtF[brow * HID + j] = hn[s][v];
      if (coutF) coutF[brow * HID + j] = cn[s][v];
      if (seqout) seqout[brow * HID + j] = (_Float16)hn[s][v];
    }
  }
  __syncthreads();
}

__global__ void lstm_kernel(const _Float16* __restrict__ X,      // [T*64][288] f16
                            const int* __restrict__ done,        // [T*64]
                            const float* __restrict__ h0g, const float* __restrict__ c0g,
                            const _Float16* __restrict__ wih0, const _Float16* __restrict__ whh0,
                            const float* __restrict__ bih0, const float* __restrict__ bhh0,
                            const _Float16* __restrict__ wih1, const _Float16* __restrict__ whh1,
                            const float* __restrict__ bih1, const float* __restrict__ bhh1,
                            _Float16* __restrict__ core_out,     // [T*64][256] f16
                            float* __restrict__ hT, float* __restrict__ cT) {
  extern __shared__ char dynsm[];
  _Float16* hsh = (_Float16*)dynsm;                 // [2][64][256] f16 = 64 KB
  float*    csh = (float*)(dynsm + 2 * BB * HID * 2); // [2][64][256] f32 = 128 KB
  const int tid = threadIdx.x;

  for (int i = tid; i < 2 * BB * HID; i += blockDim.x) {
    hsh[i] = (_Float16)h0g[i];
    csh[i] = c0g[i];
  }
  __syncthreads();

  for (int t = 0; t < TT; ++t) {
    // done-mask both layers' h and c (h *= notdone)
    for (int i = tid; i < 2 * BB * HID; i += blockDim.x) {
      int b = (i >> 8) & 63;
      if (done[t * BB + b]) { hsh[i] = (_Float16)0.0f; csh[i] = 0.0f; }
    }
    __syncthreads();

    bool last = (t == TT - 1);
    lstm_layer<FEAT>(X + (size_t)t * BB * FEAT, FEAT,
                     hsh, csh, hsh, wih0, whh0, bih0, bhh0,
                     last ? hT : nullptr, last ? cT : nullptr, nullptr);
    lstm_layer<HID>(hsh, HID,                       // layer-1 input = new h0 (in LDS)
                    hsh + BB * HID, csh + BB * HID, hsh + BB * HID,
                    wih1, whh1, bih1, bhh1,
                    last ? hT + BB * HID : nullptr, last ? cT + BB * HID : nullptr,
                    core_out + (size_t)t * BB * HID);
  }
}

// ---------------- heads: one N=16 WMMA tile covers policy(12)+baseline(1) ----------------
__global__ void heads_kernel(const _Float16* __restrict__ co, const _Float16* __restrict__ hw,
                             const float* __restrict__ hb,
                             float* __restrict__ pol, float* __restrict__ base) {
  const int wid = threadIdx.x >> 5;
  const int mt = blockIdx.x * (blockDim.x >> 5) + wid;  // 0..255 (4096 rows / 16)
  const int lane = threadIdx.x & 31, r = lane & 15, g = lane >> 4;
  v8f acc;
#pragma unroll
  for (int v = 0; v < 8; ++v) acc[v] = hb[r];
  for (int ks = 0; ks < HID / 32; ++ks) {
    v16h a = load_a16(co + (size_t)mt * 16 * HID + ks * 32, HID);
    v16h b = load_b16(hw + ks * 32, HID);
    acc = wmma16(a, b, acc);
  }
#pragma unroll
  for (int v = 0; v < 8; ++v) {
    int m = mt * 16 + v + 8 * g;
    if (r < NA)       pol[m * NA + r] = acc[v];
    else if (r == NA) base[m] = acc[v];
  }
}

__global__ void argmax_kernel(const float* __restrict__ pol, float* __restrict__ act) {
  int i = blockIdx.x * blockDim.x + threadIdx.x;
  if (i < NB) {
    const float* row = pol + i * NA;
    int bi = 0; float bv = row[0];
    for (int k = 1; k < NA; ++k) { float v = row[k]; if (v > bv) { bv = v; bi = k; } }
    act[i] = (float)bi;
  }
}

// ---------------- launcher ----------------
extern "C" void kernel_launch(void* const* d_in, const int* in_sizes, int n_in,
                              void* d_out, int out_size, void* d_ws, size_t ws_size,
                              hipStream_t stream) {
  const float* frame   = (const float*)d_in[0];
  const int*   done    = (const int*)d_in[1];
  const float* h0      = (const float*)d_in[2];
  const float* c0      = (const float*)d_in[3];
  const float* cw1     = (const float*)d_in[4];
  const float* cb1     = (const float*)d_in[5];
  const float* cw2     = (const float*)d_in[6];
  const float* cb2     = (const float*)d_in[7];
  const float* cw3     = (const float*)d_in[8];
  const float* cb3     = (const float*)d_in[9];
  const float* cw4     = (const float*)d_in[10];
  const float* cb4     = (const float*)d_in[11];
  const float* wih0    = (const float*)d_in[12];
  const float* whh0    = (const float*)d_in[13];
  const float* bih0    = (const float*)d_in[14];
  const float* bhh0    = (const float*)d_in[15];
  const float* wih1    = (const float*)d_in[16];
  const float* whh1    = (const float*)d_in[17];
  const float* bih1    = (const float*)d_in[18];
  const float* bhh1    = (const float*)d_in[19];
  const float* pw      = (const float*)d_in[20];
  const float* pb      = (const float*)d_in[21];
  const float* bw      = (const float*)d_in[22];
  const float* bbias   = (const float*)d_in[23];

  // workspace bump allocator (256B aligned)
  char* ws = (char*)d_ws;
  size_t off = 0;
  auto alloc = [&](size_t bytes) -> char* {
    char* p = ws + off;
    off += (bytes + 255) & ~(size_t)255;
    return p;
  };
  _Float16* act1    = (_Float16*)alloc((size_t)NB * 32 * 441 * 2);
  _Float16* act2    = (_Float16*)alloc((size_t)NB * 32 * 121 * 2);
  _Float16* act3    = (_Float16*)alloc((size_t)NB * 32 * 36 * 2);
  _Float16* coreinp = (_Float16*)alloc((size_t)NB * FEAT * 2);
  _Float16* coreout = (_Float16*)alloc((size_t)NB * HID * 2);
  _Float16* w2h     = (_Float16*)alloc(9216 * 2);
  _Float16* w3h     = (_Float16*)alloc(9216 * 2);
  _Float16* w4h     = (_Float16*)alloc(9216 * 2);
  _Float16* wih0h   = (_Float16*)alloc((size_t)4 * HID * FEAT * 2);
  _Float16* whh0h   = (_Float16*)alloc((size_t)4 * HID * HID * 2);
  _Float16* wih1h   = (_Float16*)alloc((size_t)4 * HID * HID * 2);
  _Float16* whh1h   = (_Float16*)alloc((size_t)4 * HID * HID * 2);
  _Float16* headw   = (_Float16*)alloc(16 * HID * 2);
  float*    headb   = (float*)alloc(16 * 4);

  float* out_pol  = (float*)d_out;                 // 4096*12
  float* out_base = out_pol + NB * NA;             // 4096
  float* out_act  = out_base + NB;                 // 4096
  float* out_hT   = out_act + NB;                  // 2*64*256
  float* out_cT   = out_hT + 2 * BB * HID;         // 2*64*256

  // weight conversions
  cvt_f16_kernel<<<(9216 + 255) / 256, 256, 0, stream>>>(cw2, w2h, 9216);
  cvt_f16_kernel<<<(9216 + 255) / 256, 256, 0, stream>>>(cw3, w3h, 9216);
  cvt_f16_kernel<<<(9216 + 255) / 256, 256, 0, stream>>>(cw4, w4h, 9216);
  cvt_f16_kernel<<<(4 * HID * FEAT + 255) / 256, 256, 0, stream>>>(wih0, wih0h, 4 * HID * FEAT);
  cvt_f16_kernel<<<(4 * HID * HID + 255) / 256, 256, 0, stream>>>(whh0, whh0h, 4 * HID * HID);
  cvt_f16_kernel<<<(4 * HID * HID + 255) / 256, 256, 0, stream>>>(wih1, wih1h, 4 * HID * HID);
  cvt_f16_kernel<<<(4 * HID * HID + 255) / 256, 256, 0, stream>>>(whh1, whh1h, 4 * HID * HID);
  head_pack_kernel<<<(16 * HID + 255) / 256, 256, 0, stream>>>(pw, pb, bw, bbias, headw, headb);

  // conv stack
  {
    const int n1 = NB * 32 * 441;
    conv1_kernel<<<(n1 + 255) / 256, 256, 0, stream>>>(frame, cw1, cb1, act1);
  }
  conv_wmma_kernel<21, 11, 128><<<NB, 256, 128 * 288 * 2, stream>>>(act1, w2h, cb2, act2);
  conv_wmma_kernel<11, 6, 48><<<NB, 256, 48 * 288 * 2, stream>>>(act2, w3h, cb3, act3);
  conv_wmma_kernel<6, 3, 16><<<NB, 256, 16 * 288 * 2, stream>>>(act3, w4h, cb4, coreinp);

  // persistent LSTM: 1 workgroup, 32 waves, 192 KB LDS (h f16 + c f32, both layers)
  lstm_kernel<<<1, 1024, 2 * BB * HID * 2 + 2 * BB * HID * 4, stream>>>(
      coreinp, done, h0, c0, wih0h, whh0h, bih0, bhh0, wih1h, whh1h, bih1, bhh1,
      coreout, out_hT, out_cT);

  // heads + argmax
  heads_kernel<<<32, 256, 0, stream>>>(coreout, headw, headb, out_pol, out_base);
  argmax_kernel<<<(NB + 255) / 256, 256, 0, stream>>>(out_pol, out_act);
}
